// HRViTAxialBlock_17978733101095
// MI455X (gfx1250) — compile-verified, matrix operand order checked by reading, not run on hardware
//
#include <hip/hip_runtime.h>
#include <cstdint>

typedef __attribute__((ext_vector_type(16))) _Float16 v16h;
typedef __attribute__((ext_vector_type(8)))  float    v8f;

#define WMMA_F32_F16(a, b, c) \
  __builtin_amdgcn_wmma_f32_16x16x32_f16(false, (a), false, (b), (short)0, (c), false, false)

// ---------------------------------------------------------------------------
// Fragment loaders (CDNA5 wave32 WMMA layouts, CDNA5 ISA §7.12.2)
// A: 16x32 f16, lane&15 = M; lanes<16 hold K 0..7 & 16..23, lanes>=16 hold 8..15 & 24..31
// B: 32x16 f16, mirrored with lane&15 = N
// C/D: 8 f32; lane&15 = N, vgpr r -> M = r (lanes<16) or 8+r (lanes>=16)
// ---------------------------------------------------------------------------
__device__ __forceinline__ v16h load_frag_a(const _Float16* base, int rs, int cs, int lane) {
  int m = lane & 15;
  int kb = (lane >> 4) * 8;
  v16h f;
#pragma unroll
  for (int i = 0; i < 8; ++i) f[i] = base[m * rs + (kb + i) * cs];
#pragma unroll
  for (int i = 0; i < 8; ++i) f[i + 8] = base[m * rs + (16 + kb + i) * cs];
  return f;
}
__device__ __forceinline__ v16h load_frag_b(const _Float16* base, int rs, int cs, int lane) {
  int n = lane & 15;
  int kb = (lane >> 4) * 8;
  v16h f;
#pragma unroll
  for (int i = 0; i < 8; ++i) f[i] = base[(kb + i) * rs + n * cs];
#pragma unroll
  for (int i = 0; i < 8; ++i) f[i + 8] = base[(16 + kb + i) * rs + n * cs];
  return f;
}

__device__ __forceinline__ float gelu_exact(float x) {
  return 0.5f * x * (1.0f + erff(x * 0.7071067811865475f));
}

// ---------------------------------------------------------------------------
// Small prep kernels
// ---------------------------------------------------------------------------
__global__ void f32_to_f16_kernel(const float* __restrict__ in, _Float16* __restrict__ out, int n) {
  int i = blockIdx.x * 256 + threadIdx.x;
  if (i < n) out[i] = (_Float16)in[i];
}

__global__ void bn_pre_kernel(const float* __restrict__ g, const float* __restrict__ b,
                              const float* __restrict__ m, const float* __restrict__ v,
                              float* __restrict__ scale, float* __restrict__ shift, int n) {
  int i = blockIdx.x * 256 + threadIdx.x;
  if (i < n) {
    float s = g[i] * rsqrtf(v[i] + 1e-5f);
    scale[i] = s;
    shift[i] = b[i] - m[i] * s;
  }
}

// ---------------------------------------------------------------------------
// LayerNorm over C of a BCHW tensor + pack f16 matrices for axial GEMM.
// block = (b*96 + h), thread = w (96 threads, coalesced over w per channel).
// Output is TOKEN-MAJOR [g][l][c] (c contiguous) so kqv B-fragments load as
// contiguous 8-half runs per lane (merge into global_load_b128):
// axisH=1: out[((b*96+w)*96 + h)*256 + c]   (g = b*96+w, l = h)
// axisH=0: out[((b*96+h)*96 + w)*256 + c]   (g = b*96+h, l = w)
// ---------------------------------------------------------------------------
__global__ void ln_pack_kernel(const float* __restrict__ in, const float* __restrict__ g,
                               const float* __restrict__ bt, _Float16* __restrict__ outp,
                               int axisH) {
  int bh = blockIdx.x;
  int b = bh / 96, h = bh % 96;
  int w = threadIdx.x;
  const float* px = in + (size_t)b * 256 * 9216 + (size_t)h * 96 + w;
  float s = 0.f, s2 = 0.f;
  for (int c = 0; c < 256; ++c) {
    float v = px[(size_t)c * 9216];
    s += v; s2 += v * v;
  }
  float mu = s * (1.0f / 256.0f);
  float var = s2 * (1.0f / 256.0f) - mu * mu;
  float r = rsqrtf(var + 1e-5f);
  _Float16* po = outp + (axisH ? (((size_t)(b * 96 + w) * 96 + h) * 256)
                               : (((size_t)(b * 96 + h) * 96 + w) * 256));
  for (int c = 0; c < 256; ++c) {
    float v = (px[(size_t)c * 9216] - mu) * r * g[c] + bt[c];
    po[c] = (_Float16)v;
  }
}

// ---------------------------------------------------------------------------
// kqv projection GEMM: per column-group g in [0, B*96): O(512x96) = W(512x256) X(256x96)
// + fused BatchNorm (scale/shift).  WMMA f16->f32.  block = 128 (4 waves),
// grid = (384, 8 Mtile-groups, 6 Ntiles).  X is token-major [g][l][c].
// ---------------------------------------------------------------------------
__global__ void kqv_gemm_kernel(const _Float16* __restrict__ w16,
                                const _Float16* __restrict__ xpack,
                                const float* __restrict__ bnscale,
                                const float* __restrict__ bnshift,
                                float* __restrict__ kqv) {
  int g = blockIdx.x;
  int lane = threadIdx.x & 31;
  int wave = threadIdx.x >> 5;
  int o0 = (blockIdx.y * 4 + wave) * 16;
  int n0 = blockIdx.z * 16;
  const _Float16* xg = xpack + (size_t)g * 96 * 256;
  v8f acc = {};
#pragma unroll
  for (int kk = 0; kk < 8; ++kk) {
    v16h a = load_frag_a(w16 + (size_t)o0 * 256 + kk * 32, 256, 1, lane);
    // elem(k=c, n=l) at xg[(n0+n)*256 + kk*32 + k]  -> rs=1, cs=256
    v16h bf = load_frag_b(xg + (size_t)n0 * 256 + kk * 32, 1, 256, lane);
    acc = WMMA_F32_F16(a, bf, acc);
  }
  int col = lane & 15;
  int rb = (lane < 16) ? 0 : 8;
  float* outp = kqv + (size_t)g * 512 * 96;
#pragma unroll
  for (int r = 0; r < 8; ++r) {
    int row = o0 + rb + r;
    outp[(size_t)row * 96 + n0 + col] = acc[r] * bnscale[row] + bnshift[row];
  }
}

// ---------------------------------------------------------------------------
// Fused axial attention (per b, column, head): logits (qk/qr/kr with BN folded
// into scale/shift) -> softmax -> (v + v_enc) apply.  All state in LDS.
// block = 96 threads (one per logit row x); grid = B*96*HEADS = 3072.
// Output: out[b*CHW + c*c_str + x*x_str + g2*g2_str]
// ---------------------------------------------------------------------------
__global__ void axial_attn_kernel(const float* __restrict__ kqv, const float* __restrict__ rel,
                                  const float* __restrict__ lsc, const float* __restrict__ lsh,
                                  float* __restrict__ out,
                                  size_t c_str, size_t x_str, size_t g2_str) {
  __shared__ float sQ[16][96];
  __shared__ float sK[16][96];
  __shared__ float sV[32][96];
  __shared__ float sRel[64][191];
  __shared__ float sL[96][97];

  int blk = blockIdx.x;
  int hd = blk & 7;
  int t = blk >> 3;
  int g2 = t % 96;
  int b = t / 96;
  int g = b * 96 + g2;
  const float* base = kqv + (size_t)g * 512 * 96 + (size_t)hd * 64 * 96;
  int tid = threadIdx.x;

  for (int i = tid; i < 16 * 96; i += 96) {
    int d = i / 96, l = i % 96;
    sK[d][l] = base[d * 96 + l];
    sQ[d][l] = base[(16 + d) * 96 + l];
  }
  for (int i = tid; i < 32 * 96; i += 96) {
    int d = i / 96, l = i % 96;
    sV[d][l] = base[(32 + d) * 96 + l];
  }
  for (int i = tid; i < 64 * 191; i += 96) sRel[i / 191][i % 191] = rel[i];
  __syncthreads();

  float s0 = lsc[hd], s1 = lsc[8 + hd], s2 = lsc[16 + hd];
  float sh = lsh[hd] + lsh[8 + hd] + lsh[16 + hd];
  int x = tid;

  float qx[16];
#pragma unroll
  for (int d = 0; d < 16; ++d) qx[d] = sQ[d][x];

  for (int y = 0; y < 96; ++y) {
    int oq = y - x + 95;   // q_enc / v_enc offset
    int ok = x - y + 95;   // k_enc offset (kr is transposed)
    float a0 = 0.f, a1 = 0.f, a2 = 0.f;
#pragma unroll
    for (int d = 0; d < 16; ++d) {
      float kd = sK[d][y];
      a0 += qx[d] * kd;
      a1 += qx[d] * sRel[d][oq];
      a2 += kd * sRel[16 + d][ok];
    }
    sL[x][y] = s0 * a0 + s1 * a1 + s2 * a2 + sh;
  }

  float mx = -1e30f;
  for (int y = 0; y < 96; ++y) mx = fmaxf(mx, sL[x][y]);
  float sum = 0.f;
  for (int y = 0; y < 96; ++y) {
    float e = __expf(sL[x][y] - mx);
    sL[x][y] = e;
    sum += e;
  }
  float inv = 1.0f / sum;

  size_t ob = (size_t)b * 256 * 9216;
  for (int d = 0; d < 32; ++d) {
    float a = 0.f;
    for (int y = 0; y < 96; ++y) a += sL[x][y] * (sV[d][y] + sRel[32 + d][y - x + 95]);
    int c = hd * 32 + d;
    out[ob + (size_t)c * c_str + (size_t)x * x_str + (size_t)g2 * g2_str] = a * inv;
  }
}

// ---------------------------------------------------------------------------
// des branch (two 16x16 matmuls + exact GELU on flat 256-element chunks) and
// residual accumulation into acc:  acc += des(x) + x.
// block = 256 threads, 16 tokens per block (16 threads per token).
// ---------------------------------------------------------------------------
__global__ void des_add_kernel(const float* __restrict__ x, const float* __restrict__ prw,
                               const float* __restrict__ prb, const float* __restrict__ plw,
                               const float* __restrict__ plb, float* __restrict__ acc) {
  __shared__ float ch[16][16][17];
  __shared__ float eg[16][16][17];
  int tid = threadIdx.x;
  int gi = tid >> 4, j = tid & 15;
  size_t tok = (size_t)blockIdx.x * 16 + gi;
  const float* px = x + tok * 256;
  for (int p = 0; p < 16; ++p) ch[gi][j][p] = px[j * 16 + p];
  __syncthreads();
  // t1[k][q] = sum_p ch[k][p]*prw[q][p] + prb[q]; eg[q][k] = gelu(t1[k][q])
  for (int k = 0; k < 16; ++k) {
    float a = prb[j];
    for (int p = 0; p < 16; ++p) a += ch[gi][k][p] * prw[j * 16 + p];
    eg[gi][j][k] = gelu_exact(a);
  }
  __syncthreads();
  // final[k2][p2] = sum_k eg[p2][k]*plw[k2][k] + plb[k2];  c = k2*16 + p2
  float* pa = acc + tok * 256;
  for (int p2 = 0; p2 < 16; ++p2) {
    float a = plb[j];
    for (int k = 0; k < 16; ++k) a += eg[gi][p2][k] * plw[j * 16 + k];
    int ci = j * 16 + p2;
    pa[ci] = pa[ci] + a + px[ci];
  }
}

// ---------------------------------------------------------------------------
// FFN LayerNorm: wave per token (contiguous 256 f32), shfl_xor reductions,
// emits f16 activation rows for the fc1 WMMA GEMM.
// ---------------------------------------------------------------------------
__global__ void ffn_ln_kernel(const float* __restrict__ acc, const float* __restrict__ g,
                              const float* __restrict__ bt, _Float16* __restrict__ act16) {
  int lane = threadIdx.x & 31;
  int wv = threadIdx.x >> 5;
  size_t tok = (size_t)blockIdx.x * 8 + wv;
  const float* p = acc + tok * 256;
  float v[8];
  float s = 0.f, s2 = 0.f;
#pragma unroll
  for (int j = 0; j < 8; ++j) {
    v[j] = p[lane + 32 * j];
    s += v[j]; s2 += v[j] * v[j];
  }
#pragma unroll
  for (int o = 16; o > 0; o >>= 1) {
    s += __shfl_xor(s, o, 32);
    s2 += __shfl_xor(s2, o, 32);
  }
  float mu = s * (1.0f / 256.0f);
  float var = s2 * (1.0f / 256.0f) - mu * mu;
  float r = rsqrtf(var + 1e-5f);
  _Float16* q = act16 + tok * 256;
#pragma unroll
  for (int j = 0; j < 8; ++j) {
    int c = lane + 32 * j;
    q[c] = (_Float16)((v[j] - mu) * r * g[c] + bt[c]);
  }
}

// ---------------------------------------------------------------------------
// fc1 GEMM: out(HID=1024 x Ntok) = W(1024x256) @ actT(256 x Ntok) + bias,
// stored f16 as (b, hid, 9216) for the depthwise conv.  LDS-staged tiles,
// block 64x32, 8 waves, full-K (256) staging.
// ---------------------------------------------------------------------------
__global__ void fc1_gemm_kernel(const _Float16* __restrict__ w16,
                                const _Float16* __restrict__ act16,
                                const float* __restrict__ bias,
                                _Float16* __restrict__ outE) {
  __shared__ _Float16 Ah[64 * 256];
  __shared__ _Float16 Bh[32 * 256];
  int tid = threadIdx.x;
  int m0 = blockIdx.x * 64;
  size_t t0 = (size_t)blockIdx.y * 32;
  const _Float16* Ag = w16 + (size_t)m0 * 256;
  const _Float16* Bg = act16 + t0 * 256;
  for (int i = tid; i < 64 * 256; i += 256) Ah[i] = Ag[i];
  for (int i = tid; i < 32 * 256; i += 256) Bh[i] = Bg[i];
  __syncthreads();
  int lane = tid & 31, wv = tid >> 5;
  int mt = wv & 3, nt = wv >> 2;
  v8f acc = {};
#pragma unroll
  for (int kk = 0; kk < 8; ++kk) {
    v16h a = load_frag_a(&Ah[(mt * 16) * 256 + kk * 32], 256, 1, lane);
    v16h b = load_frag_b(&Bh[(nt * 16) * 256 + kk * 32], 1, 256, lane);
    acc = WMMA_F32_F16(a, b, acc);
  }
  int col = lane & 15, rb = (lane < 16) ? 0 : 8;
  size_t t = t0 + nt * 16 + col;
  int bI = (int)(t / 9216);
  int n = (int)(t % 9216);
#pragma unroll
  for (int r = 0; r < 8; ++r) {
    int hid = m0 + mt * 16 + rb + r;
    float vo = acc[r] + bias[hid];
    outE[((size_t)bI * 1024 + hid) * 9216 + n] = (_Float16)vo;
  }
}

// ---------------------------------------------------------------------------
// Depthwise conv (3x3 for ch<512, 5x5 for ch>=512) + exact GELU; writes f16
// token-major (tok, hid) layout for the fc2 GEMM.
// ---------------------------------------------------------------------------
__global__ void dwconv_gelu_kernel(const _Float16* __restrict__ E, const float* __restrict__ w3,
                                   const float* __restrict__ b3, const float* __restrict__ w5,
                                   const float* __restrict__ b5, _Float16* __restrict__ g16) {
  size_t idx = (size_t)blockIdx.x * 256 + threadIdx.x;
  int w = (int)(idx % 96);
  size_t r1 = idx / 96;
  int h = (int)(r1 % 96);
  size_t r2 = r1 / 96;
  int ch = (int)(r2 % 1024);
  int b = (int)(r2 / 1024);
  const _Float16* pin = E + ((size_t)b * 1024 + ch) * 9216;
  float a;
  if (ch < 512) {
    a = b3[ch];
    const float* wt = w3 + ch * 9;
#pragma unroll
    for (int i = 0; i < 3; ++i) {
      int hh = h + i - 1;
      if ((unsigned)hh >= 96u) continue;
#pragma unroll
      for (int j = 0; j < 3; ++j) {
        int ww = w + j - 1;
        if ((unsigned)ww >= 96u) continue;
        a += (float)pin[hh * 96 + ww] * wt[i * 3 + j];
      }
    }
  } else {
    int c2 = ch - 512;
    a = b5[c2];
    const float* wt = w5 + c2 * 25;
#pragma unroll
    for (int i = 0; i < 5; ++i) {
      int hh = h + i - 2;
      if ((unsigned)hh >= 96u) continue;
#pragma unroll
      for (int j = 0; j < 5; ++j) {
        int ww = w + j - 2;
        if ((unsigned)ww >= 96u) continue;
        a += (float)pin[hh * 96 + ww] * wt[i * 5 + j];
      }
    }
  }
  a = gelu_exact(a);
  g16[((size_t)b * 9216 + (size_t)h * 96 + w) * 1024 + ch] = (_Float16)a;
}

// ---------------------------------------------------------------------------
// fc2 GEMM: out(256 x Ntok) = W(256x1024) @ gT(1024 x Ntok) + bias + res2,
// chunked-K (128) LDS staging + prefetch of next chunk, final f32 output.
// ---------------------------------------------------------------------------
__global__ void fc2_gemm_kernel(const _Float16* __restrict__ w16, const _Float16* __restrict__ g16,
                                const float* __restrict__ bias, const float* __restrict__ res2,
                                float* __restrict__ outp) {
  __shared__ _Float16 Ah[64 * 128];
  __shared__ _Float16 Bh[32 * 128];
  int tid = threadIdx.x;
  int m0 = blockIdx.x * 64;
  size_t t0 = (size_t)blockIdx.y * 32;
  int lane = tid & 31, wv = tid >> 5;
  int mt = wv & 3, nt = wv >> 2;
  v8f acc = {};
  for (int k0 = 0; k0 < 1024; k0 += 128) {
    for (int i = tid; i < 64 * 128; i += 256) {
      int r = i >> 7, c = i & 127;
      Ah[i] = w16[(size_t)(m0 + r) * 1024 + k0 + c];
    }
    for (int i = tid; i < 32 * 128; i += 256) {
      int n = i >> 7, c = i & 127;
      Bh[i] = g16[(t0 + n) * 1024 + k0 + c];
    }
    if (k0 + 128 < 1024) {
      // pull next B chunk toward the WGP while this chunk's WMMAs run
      __builtin_prefetch(&g16[(t0 + (tid >> 3)) * 1024 + k0 + 128 + ((tid & 7) << 4)], 0, 3);
    }
    __syncthreads();
#pragma unroll
    for (int kk = 0; kk < 4; ++kk) {
      v16h a = load_frag_a(&Ah[(mt * 16) * 128 + kk * 32], 128, 1, lane);
      v16h b = load_frag_b(&Bh[(nt * 16) * 128 + kk * 32], 1, 128, lane);
      acc = WMMA_F32_F16(a, b, acc);
    }
    __syncthreads();
  }
  int col = lane & 15, rb = (lane < 16) ? 0 : 8;
  size_t t = t0 + nt * 16 + col;
#pragma unroll
  for (int r = 0; r < 8; ++r) {
    int c = m0 + mt * 16 + rb + r;
    outp[t * 256 + c] = acc[r] + bias[c] + res2[t * 256 + c];
  }
}

// ---------------------------------------------------------------------------
// Host driver
// ---------------------------------------------------------------------------
extern "C" void kernel_launch(void* const* d_in, const int* in_sizes, int n_in,
                              void* d_out, int out_size, void* d_ws, size_t ws_size,
                              hipStream_t stream) {
  const float* x        = (const float*)d_in[0];
  const float* ln_g     = (const float*)d_in[1];
  const float* ln_b     = (const float*)d_in[2];
  const float* h_kqv_w  = (const float*)d_in[3];
  const float* h_bn_g   = (const float*)d_in[4];
  const float* h_bn_b   = (const float*)d_in[5];
  const float* h_bn_m   = (const float*)d_in[6];
  const float* h_bn_v   = (const float*)d_in[7];
  const float* h_lg     = (const float*)d_in[8];
  const float* h_lb     = (const float*)d_in[9];
  const float* h_lm     = (const float*)d_in[10];
  const float* h_lv     = (const float*)d_in[11];
  const float* h_rel    = (const float*)d_in[12];
  const float* w_kqv_w  = (const float*)d_in[13];
  const float* w_bn_g   = (const float*)d_in[14];
  const float* w_bn_b   = (const float*)d_in[15];
  const float* w_bn_m   = (const float*)d_in[16];
  const float* w_bn_v   = (const float*)d_in[17];
  const float* w_lg     = (const float*)d_in[18];
  const float* w_lb     = (const float*)d_in[19];
  const float* w_lm     = (const float*)d_in[20];
  const float* w_lv     = (const float*)d_in[21];
  const float* w_rel    = (const float*)d_in[22];
  const float* des_pr_w = (const float*)d_in[23];
  const float* des_pr_b = (const float*)d_in[24];
  const float* des_pl_w = (const float*)d_in[25];
  const float* des_pl_b = (const float*)d_in[26];
  const float* ffn_ln_g = (const float*)d_in[27];
  const float* ffn_ln_b = (const float*)d_in[28];
  const float* fc1_w    = (const float*)d_in[29];
  const float* fc1_b    = (const float*)d_in[30];
  const float* conv3_w  = (const float*)d_in[31];
  const float* conv3_b  = (const float*)d_in[32];
  const float* conv5_w  = (const float*)d_in[33];
  const float* conv5_b  = (const float*)d_in[34];
  const float* fc2_w    = (const float*)d_in[35];
  const float* fc2_b    = (const float*)d_in[36];
  (void)in_sizes; (void)n_in; (void)out_size; (void)ws_size;

  char* ws = (char*)d_ws;
  // workspace layout (bytes)
  _Float16* w16h   = (_Float16*)(ws + 0);          // 512*256 f16
  _Float16* w16w   = (_Float16*)(ws + 262144);
  _Float16* fc1w16 = (_Float16*)(ws + 524288);     // 1024*256 f16
  _Float16* fc2w16 = (_Float16*)(ws + 1048576);    // 256*1024 f16
  float* bnh_sc = (float*)(ws + 1572864); float* bnh_sh = bnh_sc + 512;
  float* bnw_sc = (float*)(ws + 1576960); float* bnw_sh = bnw_sc + 512;
  float* lgh_sc = (float*)(ws + 1581056); float* lgh_sh = lgh_sc + 24;
  float* lgw_sc = (float*)(ws + 1581312); float* lgw_sh = lgw_sc + 24;
  _Float16* packA = (_Float16*)(ws + 2097152);     // 18.9 MB pack / act16
  float* kqvB     = (float*)(ws + 20971520);       // 75.5 MB kqv / g16 reuse
  float* ax1C     = (float*)(ws + 96468992);       // 37.7 MB axial-1 output (BCHW)
  float* accD     = (float*)(ws + 134217728);      // 37.7 MB flat accumulator / res2
  _Float16* fc1E  = (_Float16*)(ws + 171966464);   // 75.5 MB fc1 output f16
  _Float16* g16   = (_Float16*)kqvB;               // reuse: conv+gelu output f16
  _Float16* act16 = packA;                         // reuse: FFN LN output f16

  // weight conversion + BN folding
  f32_to_f16_kernel<<<512, 256, 0, stream>>>(h_kqv_w, w16h, 131072);
  f32_to_f16_kernel<<<512, 256, 0, stream>>>(w_kqv_w, w16w, 131072);
  f32_to_f16_kernel<<<1024, 256, 0, stream>>>(fc1_w, fc1w16, 262144);
  f32_to_f16_kernel<<<1024, 256, 0, stream>>>(fc2_w, fc2w16, 262144);
  bn_pre_kernel<<<2, 256, 0, stream>>>(h_bn_g, h_bn_b, h_bn_m, h_bn_v, bnh_sc, bnh_sh, 512);
  bn_pre_kernel<<<2, 256, 0, stream>>>(w_bn_g, w_bn_b, w_bn_m, w_bn_v, bnw_sc, bnw_sh, 512);
  bn_pre_kernel<<<1, 256, 0, stream>>>(h_lg, h_lb, h_lm, h_lv, lgh_sc, lgh_sh, 24);
  bn_pre_kernel<<<1, 256, 0, stream>>>(w_lg, w_lb, w_lm, w_lv, lgw_sc, lgw_sh, 24);

  // axial attention along H
  ln_pack_kernel<<<384, 96, 0, stream>>>(x, ln_g, ln_b, packA, 1);
  kqv_gemm_kernel<<<dim3(384, 8, 6), 128, 0, stream>>>(w16h, packA, bnh_sc, bnh_sh, kqvB);
  axial_attn_kernel<<<3072, 96, 0, stream>>>(kqvB, h_rel, lgh_sc, lgh_sh, ax1C,
                                             (size_t)9216, (size_t)96, (size_t)1);
  // axial attention along W
  ln_pack_kernel<<<384, 96, 0, stream>>>(ax1C, ln_g, ln_b, packA, 0);
  kqv_gemm_kernel<<<dim3(384, 8, 6), 128, 0, stream>>>(w16w, packA, bnw_sc, bnw_sh, kqvB);
  axial_attn_kernel<<<3072, 96, 0, stream>>>(kqvB, w_rel, lgw_sc, lgw_sh, accD,
                                             (size_t)96, (size_t)1, (size_t)24576);
  // des branch + residual (flat space)
  des_add_kernel<<<2304, 256, 0, stream>>>(x, des_pr_w, des_pr_b, des_pl_w, des_pl_b, accD);
  // FFN
  ffn_ln_kernel<<<4608, 256, 0, stream>>>(accD, ffn_ln_g, ffn_ln_b, act16);
  fc1_gemm_kernel<<<dim3(16, 1152), 256, 0, stream>>>(fc1w16, act16, fc1_b, fc1E);
  dwconv_gelu_kernel<<<147456, 256, 0, stream>>>(fc1E, conv3_w, conv3_b, conv5_w, conv5_b, g16);
  fc2_gemm_kernel<<<dim3(4, 1152), 256, 0, stream>>>(fc2w16, g16, fc2_b, accD, (float*)d_out);
}
